// ImprovedGRU_59889023976073
// MI455X (gfx1250) — matrix operand order repaired and checked by reading, not compiled, not run on hardware
//
#include <hip/hip_runtime.h>
#include <hip/hip_bf16.h>
#include <math.h>

typedef __attribute__((ext_vector_type(16))) _Float16 v16h;
typedef __attribute__((ext_vector_type(8)))  float    v8f;

#define T_LEN 256
#define F1    128
#define H1    128
#define F2    128
#define H2    64
#define NB    2048
#define LDH1  136   // padded LDS row stride (halfs) for h1 state
#define LDH2  72    // padded LDS row stride (halfs) for h2 state

__device__ __forceinline__ v8f wmma16(v16h a, v16h b, v8f c) {
  // v_wmma_f32_16x16x32_f16: D = A(16x32 f16) * B(32x16 f16) + C(16x16 f32)
  return __builtin_amdgcn_wmma_f32_16x16x32_f16(false, a, false, b, (short)0, c,
                                                false, false);
}

// Fast hardware transcendentals: keep the serial gate stage short so it
// co-executes with the XDL WMMAs instead of stalling the recurrence.
__device__ __forceinline__ float fast_rcp(float x) {
#if __has_builtin(__builtin_amdgcn_rcpf)
  return __builtin_amdgcn_rcpf(x);
#else
  return 1.f / x;
#endif
}
__device__ __forceinline__ float sigmoidf(float x) {
  return fast_rcp(1.f + __expf(-x));   // v_exp + v_rcp, no IEEE div sequence
}
__device__ __forceinline__ float fast_tanh(float x) {
#if __has_builtin(__builtin_amdgcn_tanhf)
  return __builtin_amdgcn_tanhf(x);    // gfx1250 v_tanh_f32
#else
  return 1.f - 2.f * fast_rcp(1.f + __expf(2.f * x));
#endif
}

// B fragment (32x16): lane L holds column n0+L%16, K=k0+{0..7,16..23} (L<16) or k0+{8..15,24..31}.
// (one-time conversion at kernel start; stays in VGPRs for the whole t-loop)
__device__ __forceinline__ v16h load_bfrag(const float* __restrict__ B, int ldb,
                                           int k0, int n0, int lane) {
  const float* p = B + (size_t)(k0 + ((lane >> 4) << 3)) * ldb + n0 + (lane & 15);
  v16h r;
#pragma unroll
  for (int j = 0; j < 8; ++j) {
    r[j]     = (_Float16)p[(size_t)j * ldb];
    r[8 + j] = (_Float16)p[(size_t)(16 + j) * ldb];
  }
  return r;
}

// A fragment (16x32) from f16 rows: lanes 0-15 -> M=lane, K=k0+{0..7,16..23};
// lanes 16-31 -> M=lane-16, K=k0+{8..15,24..31}.  Two contiguous 16B loads.
__device__ __forceinline__ v16h load_afrag_f16(const _Float16* __restrict__ A,
                                               size_t lda, int k0, int lane) {
  const _Float16* p = A + (size_t)(lane & 15) * lda + k0 + ((lane >> 4) << 3);
  v16h r;
#pragma unroll
  for (int j = 0; j < 8; ++j) {
    r[j]     = p[j];
    r[8 + j] = p[16 + j];
  }
  return r;
}

// ---------------------------------------------------------------------------
// Pre-pass: convert x f32 -> f16 once, outside the serial recurrence.
// Pure streaming: 268 MB read + 134 MB write (~17 us at 23.3 TB/s).
// ---------------------------------------------------------------------------
__global__ __launch_bounds__(256, 1) void cvt_x_kernel(
    const float* __restrict__ x, _Float16* __restrict__ xh, size_t n /*mult of 8*/)
{
  size_t i = ((size_t)blockIdx.x * blockDim.x + threadIdx.x) * 8;
  size_t stride = (size_t)gridDim.x * blockDim.x * 8;
  for (; i < n; i += stride) {
    float4 a = *(const float4*)(x + i);
    float4 b = *(const float4*)(x + i + 4);
    _Float16 o[8] = {(_Float16)a.x, (_Float16)a.y, (_Float16)a.z, (_Float16)a.w,
                     (_Float16)b.x, (_Float16)b.y, (_Float16)b.z, (_Float16)b.w};
    *(float4*)(xh + i) = *(float4*)o;  // 16B store of 8 halfs
  }
}

// ---------------------------------------------------------------------------
// GRU layer 1 (F=128, H=128) fully fused: per-block persistent batch tile of
// 16 rows, 8 waves; wave w owns output columns [16w,16w+16) across the z/r/h
// gate groups.  Weight B-fragments live in VGPRs for the whole time loop.
// Applies BN1 and writes h_bn as f16 [B,T,128].
// ---------------------------------------------------------------------------
__global__ __launch_bounds__(256, 1) void gru1_kernel(
    const _Float16* __restrict__ xh, // [B,T,128] f16 (pre-converted)
    const float* __restrict__ W1,    // [128,384]
    const float* __restrict__ U1,    // [128,384]
    const float* __restrict__ b1,    // [2,384]
    const float* __restrict__ bng, const float* __restrict__ bnb,
    const float* __restrict__ bnm, const float* __restrict__ bnv,
    _Float16* __restrict__ hbn)      // [B,T,128] f16 out
{
  __shared__ _Float16 hlds[16 * LDH1];

  const int wave   = threadIdx.x >> 5;   // 0..7
  const int lane   = threadIdx.x & 31;
  const int batch0 = blockIdx.x * 16;
  const int c      = (wave << 4) + (lane & 15);  // owned h column
  const int nz = wave << 4, nr = 128 + nz, nh = 256 + nz;

  // Resident weight fragments: 24 x v16h = 192 VGPRs (CDNA5 wave32 budget 1024)
  v16h Wz[4], Wr[4], Wh[4], Uz[4], Ur[4], Uh[4];
#pragma unroll
  for (int kc = 0; kc < 4; ++kc) {
    Wz[kc] = load_bfrag(W1, 384, kc * 32, nz, lane);
    Wr[kc] = load_bfrag(W1, 384, kc * 32, nr, lane);
    Wh[kc] = load_bfrag(W1, 384, kc * 32, nh, lane);
    Uz[kc] = load_bfrag(U1, 384, kc * 32, nz, lane);
    Ur[kc] = load_bfrag(U1, 384, kc * 32, nr, lane);
    Uh[kc] = load_bfrag(U1, 384, kc * 32, nh, lane);
  }
  const float b0z = b1[c],       b0r = b1[128 + c],       b0h = b1[256 + c];
  const float b1z = b1[384 + c], b1r = b1[512 + c],       b1h = b1[640 + c];
  const float scale = bng[c] * rsqrtf(bnv[c] + 1e-3f);
  const float shift = bnb[c] - bnm[c] * scale;

  for (int i = threadIdx.x; i < 16 * LDH1; i += 256) hlds[i] = (_Float16)0.f;
  __syncthreads();

  v8f hprev = {};                                   // f32 state, C/D layout
  const _Float16* xbase = xh + (size_t)batch0 * T_LEN * F1;
  const size_t ldaX = (size_t)T_LEN * F1;

  for (int t = 0; t < T_LEN; ++t) {
    const _Float16* xt = xbase + (size_t)t * F1;
    if (t + 1 < T_LEN) {
      // lanes 0-15 prefetch row m start; lanes 16-31 prefetch second cacheline
      const _Float16* pf = xt + F1 + (size_t)(lane & 15) * ldaX
                              + ((lane >> 4) << 6);
      __builtin_prefetch(pf, 0, 3);
    }

    v8f accZ = {}, accR = {}, accXH = {}, accRH = {};
#pragma unroll
    for (int kc = 0; kc < 4; ++kc) {                // x_t @ W1 (K=128)
      v16h ax = load_afrag_f16(xt, ldaX, kc * 32, lane);
      accZ  = wmma16(ax, Wz[kc], accZ);
      accR  = wmma16(ax, Wr[kc], accR);
      accXH = wmma16(ax, Wh[kc], accXH);
    }
#pragma unroll
    for (int kc = 0; kc < 4; ++kc) {                // h_{t-1} @ U1 (K=128)
      v16h ah = load_afrag_f16(hlds, LDH1, kc * 32, lane);
      accZ  = wmma16(ah, Uz[kc], accZ);
      accR  = wmma16(ah, Ur[kc], accR);
      accRH = wmma16(ah, Uh[kc], accRH);
    }
    __syncthreads();  // all waves finished reading hlds for this step

    const int mrow = (lane >> 4) << 3;
    v8f hnew;
#pragma unroll
    for (int i = 0; i < 8; ++i) {
      float z  = sigmoidf(accZ[i] + b0z + b1z);
      float r  = sigmoidf(accR[i] + b0r + b1r);
      float hh = fast_tanh(accXH[i] + b0h + r * (accRH[i] + b1h)); // reset_after
      hnew[i]  = z * hprev[i] + (1.f - z) * hh;
    }
    hprev = hnew;
#pragma unroll
    for (int i = 0; i < 8; ++i) {
      int m = mrow + i;
      hlds[m * LDH1 + c] = (_Float16)hnew[i];
      float hb = hnew[i] * scale + shift;           // BN1 fused
      hbn[(((size_t)(batch0 + m)) * T_LEN + t) * H1 + c] = (_Float16)hb;
    }
    __syncthreads();  // hlds ready for next step
  }
}

// ---------------------------------------------------------------------------
// GRU layer 2 (F=128, H=64) + online-softmax attention pooling, fused.
// 4 waves / block; wave w owns h2 columns [16w,16w+16).  h2 never hits HBM.
// ---------------------------------------------------------------------------
__global__ __launch_bounds__(128, 1) void gru2_attn_kernel(
    const _Float16* __restrict__ hbn, // [B,T,128] f16 (BN'd GRU1 output)
    const float* __restrict__ W2,     // [128,192]
    const float* __restrict__ U2,     // [64,192]
    const float* __restrict__ b2,     // [2,192]
    const float* __restrict__ wa,     // [64]
    const float* __restrict__ ba,     // [1]
    float* __restrict__ ctxout)       // [B,64]
{
  __shared__ _Float16 hlds[16 * LDH2];
  __shared__ float ctx[16 * 64];
  __shared__ float srun[16], mrun[16], alpha_s[16], p_s[16];
  __shared__ float was[64];

  const int wave   = threadIdx.x >> 5;   // 0..3
  const int lane   = threadIdx.x & 31;
  const int batch0 = blockIdx.x * 16;
  const int c      = (wave << 4) + (lane & 15);
  const int nz = wave << 4, nr = 64 + nz, nh = 128 + nz;

  v16h Wz[4], Wr[4], Wh[4];  // K=128 -> 4 chunks
  v16h Uz[2], Ur[2], Uh[2];  // K=64  -> 2 chunks
#pragma unroll
  for (int kc = 0; kc < 4; ++kc) {
    Wz[kc] = load_bfrag(W2, 192, kc * 32, nz, lane);
    Wr[kc] = load_bfrag(W2, 192, kc * 32, nr, lane);
    Wh[kc] = load_bfrag(W2, 192, kc * 32, nh, lane);
  }
#pragma unroll
  for (int kc = 0; kc < 2; ++kc) {
    Uz[kc] = load_bfrag(U2, 192, kc * 32, nz, lane);
    Ur[kc] = load_bfrag(U2, 192, kc * 32, nr, lane);
    Uh[kc] = load_bfrag(U2, 192, kc * 32, nh, lane);
  }
  const float b0z = b2[c],       b0r = b2[64 + c],  b0h = b2[128 + c];
  const float b1z = b2[192 + c], b1r = b2[256 + c], b1h = b2[320 + c];

  for (int i = threadIdx.x; i < 16 * LDH2; i += 128) hlds[i] = (_Float16)0.f;
  for (int i = threadIdx.x; i < 16 * 64; i += 128) ctx[i] = 0.f;
  if (threadIdx.x < 64) was[threadIdx.x] = wa[threadIdx.x];
  if (threadIdx.x < 16) { srun[threadIdx.x] = 0.f; mrun[threadIdx.x] = -1e30f; }
  __syncthreads();

  v8f hprev = {};
  const _Float16* xb = hbn + (size_t)batch0 * T_LEN * F2;
  const size_t ldaX  = (size_t)T_LEN * F2;
  const float bav = ba[0];

  for (int t = 0; t < T_LEN; ++t) {
    const _Float16* xt = xb + (size_t)t * F2;
    if (t + 1 < T_LEN) {
      const _Float16* pf = xt + F2 + (size_t)(lane & 15) * ldaX
                              + ((lane >> 4) << 6);
      __builtin_prefetch(pf, 0, 3);
    }
    v8f accZ = {}, accR = {}, accXH = {}, accRH = {};
#pragma unroll
    for (int kc = 0; kc < 4; ++kc) {                // h1_bn @ W2
      v16h ax = load_afrag_f16(xt, ldaX, kc * 32, lane);
      accZ  = wmma16(ax, Wz[kc], accZ);
      accR  = wmma16(ax, Wr[kc], accR);
      accXH = wmma16(ax, Wh[kc], accXH);
    }
#pragma unroll
    for (int kc = 0; kc < 2; ++kc) {                // h2_{t-1} @ U2
      v16h ah = load_afrag_f16(hlds, LDH2, kc * 32, lane);
      accZ  = wmma16(ah, Uz[kc], accZ);
      accR  = wmma16(ah, Ur[kc], accR);
      accRH = wmma16(ah, Uh[kc], accRH);
    }
    __syncthreads();

    const int mrow = (lane >> 4) << 3;
    v8f hnew;
#pragma unroll
    for (int i = 0; i < 8; ++i) {
      float z  = sigmoidf(accZ[i] + b0z + b1z);
      float r  = sigmoidf(accR[i] + b0r + b1r);
      float hh = fast_tanh(accXH[i] + b0h + r * (accRH[i] + b1h));
      hnew[i]  = z * hprev[i] + (1.f - z) * hh;
    }
    hprev = hnew;
#pragma unroll
    for (int i = 0; i < 8; ++i)
      hlds[(mrow + i) * LDH2 + c] = (_Float16)hnew[i];
    __syncthreads();

    // attention logit + online softmax bookkeeping (one thread per batch row)
    if (threadIdx.x < 16) {
      int row = threadIdx.x;
      float d = 0.f;
#pragma unroll 8
      for (int j = 0; j < 64; ++j) d += (float)hlds[row * LDH2 + j] * was[j];
      float e  = fast_tanh(d + bav);
      float mo = mrun[row];
      float mn = fmaxf(mo, e);
      float al = __expf(mo - mn);   // ==0 on first step (mo=-1e30)
      float p  = __expf(e - mn);
      srun[row]    = srun[row] * al + p;
      mrun[row]    = mn;
      alpha_s[row] = al;
      p_s[row]     = p;
    }
    __syncthreads();
    // ctx = ctx*alpha + p*h2_t   (1024 elems / 128 threads)
#pragma unroll
    for (int i = 0; i < 8; ++i) {
      int idx = threadIdx.x * 8 + i;
      int row = idx >> 6, col = idx & 63;
      ctx[idx] = ctx[idx] * alpha_s[row] + p_s[row] * (float)hlds[row * LDH2 + col];
    }
    __syncthreads();
  }

#pragma unroll
  for (int i = 0; i < 8; ++i) {
    int idx = threadIdx.x * 8 + i;
    int row = idx >> 6, col = idx & 63;
    ctxout[(size_t)(batch0 + row) * H2 + col] = ctx[idx] / srun[row];
  }
}

// ---------------------------------------------------------------------------
// Classifier: relu->BN->relu->BN->sigmoid.  67 MFLOP total: plain VALU.
// One block (128 threads) per batch row.
// ---------------------------------------------------------------------------
__global__ __launch_bounds__(128, 1) void cls_kernel(
    const float* __restrict__ ctx,  // [B,64]
    const float* __restrict__ c1w, const float* __restrict__ c1b,
    const float* __restrict__ g1, const float* __restrict__ be1,
    const float* __restrict__ mu1, const float* __restrict__ v1,
    const float* __restrict__ c2w, const float* __restrict__ c2b,
    const float* __restrict__ g2, const float* __restrict__ be2,
    const float* __restrict__ mu2, const float* __restrict__ v2,
    const float* __restrict__ c3w, const float* __restrict__ c3b,
    float* __restrict__ out)        // [B]
{
  __shared__ float xr[64], y1[128], y2[64];
  const int row = blockIdx.x;
  const int tid = threadIdx.x;
  if (tid < 64) xr[tid] = ctx[(size_t)row * 64 + tid];
  __syncthreads();
  {
    float a = 0.f;
#pragma unroll 8
    for (int k = 0; k < 64; ++k) a += xr[k] * c1w[k * 128 + tid];
    a = fmaxf(a + c1b[tid], 0.f);
    float sc = g1[tid] * rsqrtf(v1[tid] + 1e-3f);
    y1[tid] = (a - mu1[tid]) * sc + be1[tid];
  }
  __syncthreads();
  if (tid < 64) {
    float a = 0.f;
#pragma unroll 8
    for (int k = 0; k < 128; ++k) a += y1[k] * c2w[k * 64 + tid];
    a = fmaxf(a + c2b[tid], 0.f);
    float sc = g2[tid] * rsqrtf(v2[tid] + 1e-3f);
    y2[tid] = (a - mu2[tid]) * sc + be2[tid];
  }
  __syncthreads();
  if (tid == 0) {
    float a = 0.f;
    for (int k = 0; k < 64; ++k) a += y2[k] * c3w[k];
    out[row] = sigmoidf(a + c3b[0]);
  }
}

// ---------------------------------------------------------------------------
extern "C" void kernel_launch(void* const* d_in, const int* in_sizes, int n_in,
                              void* d_out, int out_size, void* d_ws, size_t ws_size,
                              hipStream_t stream) {
  (void)in_sizes; (void)n_in; (void)out_size; (void)ws_size;
  const float* x    = (const float*)d_in[0];
  const float* W1   = (const float*)d_in[1];
  const float* U1   = (const float*)d_in[2];
  const float* b1   = (const float*)d_in[3];
  const float* bn1g = (const float*)d_in[4];
  const float* bn1b = (const float*)d_in[5];
  const float* bn1m = (const float*)d_in[6];
  const float* bn1v = (const float*)d_in[7];
  const float* W2   = (const float*)d_in[8];
  const float* U2   = (const float*)d_in[9];
  const float* b2   = (const float*)d_in[10];
  const float* wa   = (const float*)d_in[11];
  const float* ba   = (const float*)d_in[12];
  const float* c1w  = (const float*)d_in[13];
  const float* c1b  = (const float*)d_in[14];
  const float* g1   = (const float*)d_in[15];
  const float* be1  = (const float*)d_in[16];
  const float* mu1  = (const float*)d_in[17];
  const float* v1   = (const float*)d_in[18];
  const float* c2w  = (const float*)d_in[19];
  const float* c2b  = (const float*)d_in[20];
  const float* g2   = (const float*)d_in[21];
  const float* be2  = (const float*)d_in[22];
  const float* mu2  = (const float*)d_in[23];
  const float* v2   = (const float*)d_in[24];
  const float* c3w  = (const float*)d_in[25];
  const float* c3b  = (const float*)d_in[26];
  float* out = (float*)d_out;

  // Workspace layout:
  //   [0)              x f16   [2048,256,128]  128 MiB
  //   [xbytes)         h1_bn   [2048,256,128]  128 MiB (f16)
  //   [xbytes+hbytes)  ctx f32 [2048,64]       0.5 MiB
  const size_t xelems = (size_t)NB * T_LEN * F1;
  _Float16* xhalf = (_Float16*)d_ws;
  _Float16* hbn = (_Float16*)((char*)d_ws + xelems * sizeof(_Float16));
  float* ctx = (float*)((char*)d_ws + 2 * xelems * sizeof(_Float16));

  cvt_x_kernel<<<4096, 256, 0, stream>>>(x, xhalf, xelems);
  gru1_kernel<<<NB / 16, 256, 0, stream>>>(xhalf, W1, U1, b1, bn1g, bn1b, bn1m,
                                           bn1v, hbn);
  gru2_attn_kernel<<<NB / 16, 128, 0, stream>>>(hbn, W2, U2, b2, wa, ba, ctx);
  cls_kernel<<<NB, 128, 0, stream>>>(ctx, c1w, c1b, g1, be1, mu1, v1,
                                     c2w, c2b, g2, be2, mu2, v2, c3w, c3b, out);
}
